// LocalGeometryLoss_45603962749037
// MI455X (gfx1250) — matrix operand order repaired and checked by reading, not compiled, not run on hardware
//
#include <hip/hip_runtime.h>
#include <hip/hip_bf16.h>

// ---------------------------------------------------------------------------
// LocalGeometryLoss for MI455X (gfx1250, wave32, WMMA).
// Phase A: bf16 WMMA similarity GEMM (4096x8192x768) fused with per-row top-5.
//          4-way N-blocking: each A fragment feeds 4 WMMAs (ds/wmma = 0.5).
// Phase B: 20480 sparse pair d2 evaluations + deterministic reduction.
// ---------------------------------------------------------------------------

#define BCUR  4096
#define BPREV 8192
#define DIM   768
#define KNB   5
#define NPAIR (BCUR * KNB)           // 20480
#define LDSA_STRIDE 776              // 768 + 8 halfwords pad -> conflict-free b128
#define NBLK  4                      // column tiles per wave iteration

#if defined(__has_builtin)
# if __has_builtin(__builtin_amdgcn_global_load_async_to_lds_b128)
#  define LGL_ASYNC_LDS 1
# endif
#endif

typedef __attribute__((ext_vector_type(8)))  float  v8f;
typedef __attribute__((ext_vector_type(16))) __bf16 v16bf;
typedef __attribute__((ext_vector_type(4)))  int    v4i;

// HIP prints AS(1) as __device__ and AS(3) as __shared__ in diagnostics;
// the async-LDS builtin takes typed v4i pointers in those address spaces.
typedef __attribute__((address_space(1))) v4i* as1_v4i_p;
typedef __attribute__((address_space(3))) v4i* as3_v4i_p;

union V16BF {
    v16bf v;
    uint4 q[2];
};

__device__ __forceinline__ unsigned short f32_to_bf16_rne(float f) {
    unsigned int u = __float_as_uint(f);
    unsigned int r = u + 0x7FFFu + ((u >> 16) & 1u);   // round-to-nearest-even
    return (unsigned short)(r >> 16);
}

// ---------------- K1a: normalize hidden_previous rows -> bf16 ---------------
__global__ void lgl_norm_prev(const float* __restrict__ x,
                              unsigned short* __restrict__ hp) {
    const int row = blockIdx.x;
    const int tid = threadIdx.x;                        // 256 threads
    const float* xr = x + (size_t)row * DIM;
    __shared__ float red[256];
    float s = 0.f;
    for (int k = tid; k < DIM; k += 256) { float v = xr[k]; s += v * v; }
    red[tid] = s;
    __syncthreads();
    for (int off = 128; off > 0; off >>= 1) {
        if (tid < off) red[tid] += red[tid + off];
        __syncthreads();
    }
    const float scale = 1.0f / fmaxf(sqrtf(red[0]), 1e-12f);
    unsigned short* out = hp + (size_t)row * DIM;
    for (int k = tid; k < DIM; k += 256)
        out[k] = f32_to_bf16_rne(xr[k] * scale);
}

// ------------- K1b: normalize hidden_current rows -> f32 + sq ---------------
__global__ void lgl_norm_cur(const float* __restrict__ x,
                             float* __restrict__ hc,
                             float* __restrict__ sq) {
    const int row = blockIdx.x;
    const int tid = threadIdx.x;                        // 256 threads
    const float* xr = x + (size_t)row * DIM;
    __shared__ float red[256];
    float s = 0.f;
    for (int k = tid; k < DIM; k += 256) { float v = xr[k]; s += v * v; }
    red[tid] = s;
    __syncthreads();
    for (int off = 128; off > 0; off >>= 1) {
        if (tid < off) red[tid] += red[tid + off];
        __syncthreads();
    }
    const float nrm   = fmaxf(sqrtf(red[0]), 1e-12f);
    const float scale = 1.0f / nrm;
    float* out = hc + (size_t)row * DIM;
    for (int k = tid; k < DIM; k += 256) out[k] = xr[k] * scale;
    if (tid == 0) sq[row] = red[0] * scale * scale;     // == sum(hc*hc)
}

__device__ __forceinline__ void topk_insert(float (&val)[KNB], int (&idx)[KNB],
                                            float v, int j) {
    if (v <= val[KNB - 1]) return;
    val[KNB - 1] = v; idx[KNB - 1] = j;
#pragma unroll
    for (int k = KNB - 1; k > 0; --k) {
        if (val[k] > val[k - 1]) {
            float tv = val[k]; val[k] = val[k - 1]; val[k - 1] = tv;
            int   ti = idx[k]; idx[k] = idx[k - 1]; idx[k - 1] = ti;
        }
    }
}

// ---- K2: WMMA similarity GEMM fused with per-row top-5 neighbor search -----
// grid = 256 (16-row strips of prev[:4096]), block = 128 (4 waves).
// Each wave accumulates NBLK=4 16x16 tiles per iteration, reusing A fragments.
__global__ __launch_bounds__(128) void lgl_topk_wmma(
        const unsigned short* __restrict__ hp,      // bf16 bits, [8192][768]
        const int* __restrict__ labels_prev,
        int* __restrict__ pj, float* __restrict__ pe) {

    __shared__ unsigned short Abuf[16 * LDSA_STRIDE];
    __shared__ float tile[4][16][NBLK * 16 + 4];
    __shared__ float mval[16][4][KNB];
    __shared__ int   midx[16][4][KNB];

    const int tid  = threadIdx.x;
    const int wave = tid >> 5;
    const int lane = tid & 31;
    const int half = lane >> 4;
    const int m    = lane & 15;
    const int i0   = blockIdx.x * 16;

    // Stage 16 rows of A (bf16) into padded LDS via b128 copies.
    // CDNA5 path: GLOBAL_LOAD_ASYNC_TO_LDS_B128 (ASYNCcnt), else plain b128.
    for (int t = tid; t < 16 * (DIM / 8); t += 128) {
        int r = t / (DIM / 8), q = t % (DIM / 8);
        const uint4* src = ((const uint4*)(hp + (size_t)(i0 + r) * DIM)) + q;
        uint4*       dst = ((uint4*)(Abuf + r * LDSA_STRIDE)) + q;
#ifdef LGL_ASYNC_LDS
        __builtin_amdgcn_global_load_async_to_lds_b128(
            (as1_v4i_p)src, (as3_v4i_p)dst, 0, 0);
#else
        *dst = *src;
#endif
    }
#ifdef LGL_ASYNC_LDS
# if __has_builtin(__builtin_amdgcn_s_wait_asynccnt)
    __builtin_amdgcn_s_wait_asynccnt(0);
# else
    asm volatile("s_wait_asynccnt 0x0" ::: "memory");
# endif
#endif
    __syncthreads();

    float val[KNB]; int idx[KNB];
#pragma unroll
    for (int k = 0; k < KNB; ++k) { val[k] = -3.4e38f; idx[k] = -1; }

    // A fragment base (ISA 16-bit A layout: two 16B chunks, K-half by lane>>4)
    const unsigned short* Arow = Abuf + m * LDSA_STRIDE + 8 * half;

    const int NCT = BPREV / 16;                          // 512 column tiles
    for (int ctb = wave * NBLK; ctb < NCT; ctb += 4 * NBLK) {
        const int col0 = ctb * 16;

        // B fragment bases (ISA 16-bit B layout: two 16B chunks, K-half by lane>>4)
        const unsigned short* Brow[NBLK];
#pragma unroll
        for (int t = 0; t < NBLK; ++t)
            Brow[t] = hp + (size_t)(col0 + t * 16 + m) * DIM + 8 * half;

        if (ctb + 4 * NBLK < NCT) {                      // global_prefetch_b8
#pragma unroll
            for (int t = 0; t < NBLK; ++t)
                __builtin_prefetch(
                    hp + (size_t)((ctb + 4 * NBLK + t) * 16 + m) * DIM, 0, 1);
        }

        v8f acc[NBLK] = {};
#pragma unroll
        for (int kb = 0; kb < DIM; kb += 32) {
            V16BF a;
            a.q[0] = *(const uint4*)(Arow + kb);
            a.q[1] = *(const uint4*)(Arow + kb + 16);
#pragma unroll
            for (int t = 0; t < NBLK; ++t) {
                V16BF b;
                b.q[0] = *(const uint4*)(Brow[t] + kb);
                b.q[1] = *(const uint4*)(Brow[t] + kb + 16);
                acc[t] = __builtin_amdgcn_wmma_f32_16x16x32_bf16(
                        false, a.v, false, b.v, (short)0, acc[t], false, false);
            }
        }

        // Spill 4 16x16 f32 tiles to LDS (C layout: VGPR r -> row r + 8*half).
#pragma unroll
        for (int t = 0; t < NBLK; ++t)
#pragma unroll
            for (int r = 0; r < 8; ++r)
                tile[wave][r + 8 * half][t * 16 + m] = acc[t][r];
        __builtin_amdgcn_wave_barrier();                  // DS in-order per wave

        if (lane < 16) {
            const int ig = i0 + lane;
            for (int cc = 0; cc < NBLK * 16; ++cc) {
                const int jg = col0 + cc;
                if (jg != ig)
                    topk_insert(val, idx, tile[wave][lane][cc], jg);
            }
        }
        __builtin_amdgcn_wave_barrier();                  // WAR vs next tiles
    }

    if (lane < 16) {
#pragma unroll
        for (int k = 0; k < KNB; ++k) {
            mval[lane][wave][k] = val[k];
            midx[lane][wave][k] = idx[k];
        }
    }
    __syncthreads();

    // Merge 4 per-wave lists and emit neighbor slots (deterministic layout).
    if (tid < 16) {
        float fv[KNB]; int fi[KNB];
#pragma unroll
        for (int k = 0; k < KNB; ++k) { fv[k] = -3.4e38f; fi[k] = -1; }
        for (int w = 0; w < 4; ++w)
#pragma unroll
            for (int k = 0; k < KNB; ++k)
                topk_insert(fv, fi, mval[tid][w][k], midx[tid][w][k]);

        const int ig = i0 + tid;
        const int li = labels_prev[ig];
#pragma unroll
        for (int n = 0; n < KNB; ++n) {
            const int j = fi[n];
            float e = 0.f;
            if (j >= 0 && j < BCUR)
                e = (labels_prev[j] == li) ? 1.f : -1.f;
            pj[(size_t)ig * KNB + n] = j;
            pe[(size_t)ig * KNB + n] = e;
        }
    }
}

// -------- K3: per-pair d2 contribution (zero for masked slots) --------------
__global__ void lgl_pair_d2(const float* __restrict__ hc,
                            const float* __restrict__ sq,
                            const int* __restrict__ pj,
                            const float* __restrict__ pe,
                            float* __restrict__ contrib) {
    const int b   = blockIdx.x;
    const int tid = threadIdx.x;                        // 128 threads
    const int i   = b / KNB;
    const int j   = pj[b];
    const float e = pe[b];
    __shared__ float red[128];
    float s = 0.f;
    if (e != 0.f) {
        const float* xi = hc + (size_t)i * DIM;
        const float* xj = hc + (size_t)j * DIM;
        for (int k = tid; k < DIM; k += 128) s += xi[k] * xj[k];
    }
    red[tid] = s;
    __syncthreads();
    for (int off = 64; off > 0; off >>= 1) {
        if (tid < off) red[tid] += red[tid + off];
        __syncthreads();
    }
    if (tid == 0) {
        float out = 0.f;
        if (e != 0.f) {
            const float d2 = fmaxf(sq[i] + sq[j] - 2.0f * red[0], 0.0f);
            out = e * d2;
        }
        contrib[b] = out;
    }
}

// -------- K4: fixed-order deterministic reduction + final scaling -----------
__global__ void lgl_reduce(const float* __restrict__ contrib,
                           float* __restrict__ out) {
    const int tid = threadIdx.x;                        // 256 threads, 1 block
    __shared__ float red[256];
    float s = 0.f;
    for (int t = tid; t < NPAIR; t += 256) s += contrib[t];
    red[tid] = s;
    __syncthreads();
    for (int off = 128; off > 0; off >>= 1) {
        if (tid < off) red[tid] += red[tid + off];
        __syncthreads();
    }
    if (tid == 0)
        out[0] = 0.5f * red[0] / ((float)BCUR * (float)BCUR);
}

// ---------------------------------------------------------------------------
extern "C" void kernel_launch(void* const* d_in, const int* in_sizes, int n_in,
                              void* d_out, int out_size, void* d_ws, size_t ws_size,
                              hipStream_t stream) {
    const float* hid_cur  = (const float*)d_in[0];      // [4096][768] f32
    const float* hid_prev = (const float*)d_in[1];      // [8192][768] f32
    const int*   lab_prev = (const int*)d_in[3];        // [8192]

    char* ws = (char*)d_ws;
    size_t off = 0;
    unsigned short* hp = (unsigned short*)(ws + off); off += (size_t)BPREV * DIM * 2;  // 12.0 MB
    float* hc          = (float*)(ws + off);          off += (size_t)BCUR  * DIM * 4;  // 12.0 MB
    float* sq          = (float*)(ws + off);          off += (size_t)BCUR * 4;
    int*   pj          = (int*)(ws + off);            off += (size_t)NPAIR * 4;
    float* pe          = (float*)(ws + off);          off += (size_t)NPAIR * 4;
    float* contrib     = (float*)(ws + off);          off += (size_t)NPAIR * 4;

    lgl_norm_prev<<<BPREV, 256, 0, stream>>>(hid_prev, hp);
    lgl_norm_cur <<<BCUR,  256, 0, stream>>>(hid_cur, hc, sq);
    lgl_topk_wmma<<<BCUR / 16, 128, 0, stream>>>(hp, lab_prev, pj, pe);
    lgl_pair_d2  <<<NPAIR, 128, 0, stream>>>(hc, sq, pj, pe, contrib);
    lgl_reduce   <<<1, 256, 0, stream>>>(contrib, (float*)d_out);
}
